// PreLNTransformerEncoderBlock_18141941859051
// MI455X (gfx1250) — matrix operand order repaired and checked by reading, not compiled, not run on hardware
//
#include <hip/hip_runtime.h>

// ---------------------------------------------------------------------------
// Pre-LN transformer encoder block for MI455X (gfx1250, wave32, WMMA).
// All matmuls on v_wmma_f32_16x16x32_bf16 (bf16 operands, f32 accumulate).
// GEMM tiles staged by the Tensor Data Mover (double-buffered, TENSORcnt-
// pipelined); attention V tiles staged with async global->LDS; flash-style
// online softmax with a single cross-half shuffle per statistic.
// ---------------------------------------------------------------------------

#define D_MODEL 1024
#define FF_DIM  4096
#define SEQ     2048
#define BATCH   4
#define HEADS   16
#define DK      64
#define ROWS    (BATCH * SEQ)   // 8192

// ---- gfx1250 feature probes (guarded: fallbacks keep the compile green) ----
#if defined(__has_builtin)
#if __has_builtin(__builtin_amdgcn_tensor_load_to_lds)
#define HAS_TDM 1
#endif
#if __has_builtin(__builtin_amdgcn_global_load_async_to_lds_b128)
#define HAS_ASYNC 1
#endif
#if __has_builtin(__builtin_amdgcn_s_wait_asynccnt)
#define WAIT_ASYNC() __builtin_amdgcn_s_wait_asynccnt(0)
#endif
#endif
#ifndef HAS_TDM
#define HAS_TDM 0
#endif
#ifndef HAS_ASYNC
#define HAS_ASYNC 0
#endif
#ifndef WAIT_ASYNC
#define WAIT_ASYNC() asm volatile("s_wait_asynccnt 0" ::: "memory")
#endif

#define WAIT_DS() asm volatile("s_wait_dscnt 0" ::: "memory")

typedef __attribute__((ext_vector_type(16))) __bf16 v16bf;
typedef __attribute__((ext_vector_type(8)))  float  v8f;

union Frag {
  uint4 u4[2];
  unsigned short s[16];
  v16bf v;
};

__device__ __forceinline__ unsigned short f2bf(float f) {
  unsigned u = __float_as_uint(f);
  u += 0x7FFFu + ((u >> 16) & 1u);      // round-to-nearest-even
  return (unsigned short)(u >> 16);
}

__device__ __forceinline__ unsigned pack2(float a, float b) {
  return (unsigned)f2bf(a) | ((unsigned)f2bf(b) << 16);
}

__device__ __forceinline__ v8f wmma_bf16(v16bf a, v16bf b, v8f c) {
  return __builtin_amdgcn_wmma_f32_16x16x32_bf16(false, a, false, b, (short)0, c,
                                                 false, false);
}

#if HAS_ASYNC
// Builtin expects typed pointers: int4 in AS1 (global) / AS3 (LDS).
typedef __attribute__((ext_vector_type(4))) int async_v4i;
typedef __attribute__((address_space(1))) async_v4i* glb_b128_t;
typedef __attribute__((address_space(3))) async_v4i* lds_b128_t;
__device__ __forceinline__ glb_b128_t to_glb(const void* p) {
  return (glb_b128_t)(unsigned long long)(size_t)p;
}
__device__ __forceinline__ lds_b128_t to_lds(void* p) {
  return (lds_b128_t)(unsigned)(size_t)p;   // generic LDS addr low 32 = LDS offset
}
#endif

#if HAS_TDM
typedef __attribute__((ext_vector_type(4))) unsigned int u32x4_t;
typedef __attribute__((ext_vector_type(8))) int i32x8_t;
typedef __attribute__((ext_vector_type(4))) int i32x4_t;

// 2D TDM tile load: bf16 tile (tile_w x tile_h) from a row-major tensor with
// row stride `stride_elems`, written contiguously (optionally LDS-padded).
__device__ __forceinline__ void tdm_load_2d(const unsigned short* gsrc,
                                            unsigned short* ldst,
                                            unsigned tensor_w, unsigned stride_elems,
                                            unsigned tile_w, unsigned tile_h,
                                            unsigned pad_en, unsigned pad_interval,
                                            unsigned pad_amount) {
  const unsigned long long ga = (unsigned long long)(size_t)gsrc;
  const unsigned la = (unsigned)(size_t)ldst;
  const unsigned tensor_h = 1u << 20;
  u32x4_t g0;
  g0[0] = 1u;                                                   // count=1, user D#
  g0[1] = la;                                                   // lds_addr
  g0[2] = (unsigned)ga;                                         // global_addr[31:0]
  g0[3] = (unsigned)((ga >> 32) & 0x01FFFFFFu) | 0x80000000u;   // addr[56:32]|type=2
  i32x8_t g1;
  g1[0] = (int)((1u << 16) | (pad_en << 20) | (pad_interval << 22) |
                (pad_amount << 25));                            // data_size=2B
  g1[1] = (int)((tensor_w & 0xFFFFu) << 16);                    // tensor_dim0 lo
  g1[2] = (int)(((tensor_w >> 16) & 0xFFFFu) | ((tensor_h & 0xFFFFu) << 16));
  g1[3] = (int)(((tensor_h >> 16) & 0xFFFFu) | ((tile_w & 0xFFFFu) << 16));
  g1[4] = (int)(tile_h & 0xFFFFu);                              // tile_dim1, dim2=0
  g1[5] = (int)stride_elems;                                    // dim0_stride lo32
  g1[6] = 0;
  g1[7] = 0;
  const i32x4_t z4 = {0, 0, 0, 0};
#if defined(__clang_major__) && (__clang_major__ >= 23)
  const i32x8_t z8 = {0, 0, 0, 0, 0, 0, 0, 0};
  __builtin_amdgcn_tensor_load_to_lds(g0, g1, z4, z4, z8, 0);
#else
  __builtin_amdgcn_tensor_load_to_lds(g0, g1, z4, z4, 0);
#endif
}
#endif  // HAS_TDM

// ---------------------------------------------------------------------------
// fp32 -> bf16 elementwise (weight conversion)
// ---------------------------------------------------------------------------
__global__ __launch_bounds__(256) void cvt_f32_bf16(const float* __restrict__ in,
                                                    unsigned short* __restrict__ out,
                                                    int n) {
  int i = blockIdx.x * 256 + threadIdx.x;
  if (i < n) out[i] = f2bf(in[i]);
}

// ---------------------------------------------------------------------------
// LayerNorm over last dim (1024), one row per 256-thread block, bf16 out.
// ---------------------------------------------------------------------------
__global__ __launch_bounds__(256) void layernorm_bf16(const float* __restrict__ x,
                                                      const float* __restrict__ gamma,
                                                      const float* __restrict__ beta,
                                                      unsigned short* __restrict__ out) {
  const int row = blockIdx.x;
  const int tid = threadIdx.x;
  const float4 v = ((const float4*)(x + (size_t)row * D_MODEL))[tid];

  __shared__ float rs[256];
  __shared__ float rq[256];
  rs[tid] = v.x + v.y + v.z + v.w;
  rq[tid] = v.x * v.x + v.y * v.y + v.z * v.z + v.w * v.w;
  __syncthreads();
#pragma unroll
  for (int st = 128; st > 0; st >>= 1) {
    if (tid < st) { rs[tid] += rs[tid + st]; rq[tid] += rq[tid + st]; }
    __syncthreads();
  }
  const float mu  = rs[0] * (1.0f / D_MODEL);
  const float var = rq[0] * (1.0f / D_MODEL) - mu * mu;
  const float inv = rsqrtf(var + 1e-5f);

  const float4 g4 = ((const float4*)gamma)[tid];
  const float4 b4 = ((const float4*)beta)[tid];
  unsigned short* o = out + (size_t)row * D_MODEL + tid * 4;
  o[0] = f2bf((v.x - mu) * inv * g4.x + b4.x);
  o[1] = f2bf((v.y - mu) * inv * g4.y + b4.y);
  o[2] = f2bf((v.z - mu) * inv * g4.z + b4.z);
  o[3] = f2bf((v.w - mu) * inv * g4.w + b4.w);
}

// ---------------------------------------------------------------------------
// WMMA GEMM: C[M,N] = act(A[M,K] @ B[K,N] + bias (+ residual)).
// 128 threads = 4 waves; block tile 128x64; wave tile 32x64 (8 WMMA / K-step).
// Tiles staged by TDM (double-buffered, s_wait_tensorcnt-pipelined) when
// available; manual b128 staging otherwise. A padded to 80B rows in LDS so
// A-fragment ds_load_b128 reads stay aligned / conflict-light; B row-major.
// ---------------------------------------------------------------------------
template <bool RELU, bool RES, bool OUTBF>
__global__ __launch_bounds__(128) void gemm_bias(const unsigned short* __restrict__ A,
                                                 const unsigned short* __restrict__ Bm,
                                                 const float* __restrict__ bias,
                                                 const float* __restrict__ residual,
                                                 void* __restrict__ out,
                                                 int M, int N, int K) {
  __shared__ unsigned short As[2][128 * 40];   // [m][k], 80B row stride
  __shared__ unsigned short Bs[2][32 * 64];    // [k][n] row-major

  const int tid  = threadIdx.x;
  const int w    = tid >> 5;
  const int lane = tid & 31;
  const int g    = lane >> 4;
  const int ln   = lane & 15;
  const int m0   = blockIdx.y * 128;
  const int n0   = blockIdx.x * 64;
  const int steps = K >> 5;
  (void)M;

  v8f acc[2][4];
#pragma unroll
  for (int mi = 0; mi < 2; ++mi)
#pragma unroll
    for (int nt = 0; nt < 4; ++nt) acc[mi][nt] = {};

#if HAS_TDM
  if (tid < 32) {   // one wave drives the DMA engine
    tdm_load_2d(A + (size_t)m0 * K, As[0], K, K, 32, 128, 1, 3, 3);
    tdm_load_2d(Bm + n0, Bs[0], N, N, 64, 32, 0, 0, 0);
  }
#endif

  for (int kk = 0; kk < steps; ++kk) {
    const int cur = kk & 1;
    const int k0  = kk << 5;
#if HAS_TDM
    if (tid < 32) {
      if (kk + 1 < steps) {
        const int k1 = k0 + 32;
        tdm_load_2d(A + (size_t)m0 * K + k1, As[cur ^ 1], K, K, 32, 128, 1, 3, 3);
        tdm_load_2d(Bm + (size_t)k1 * N + n0, Bs[cur ^ 1], N, N, 64, 32, 0, 0, 0);
        __builtin_amdgcn_s_wait_tensorcnt(2);   // current pair done, next in flight
      } else {
        __builtin_amdgcn_s_wait_tensorcnt(0);
      }
    }
#else
#pragma unroll
    for (int i = 0; i < 4; ++i) {               // A: 512 b128 over 128 threads
      const int q = tid + i * 128;
      const int row = q >> 2, cv = q & 3;
      *(uint4*)&As[cur][row * 40 + cv * 8] =
          *(const uint4*)(A + (size_t)(m0 + row) * K + k0 + cv * 8);
    }
#pragma unroll
    for (int i = 0; i < 2; ++i) {               // B: 256 b128 over 128 threads
      const int q = tid + i * 128;
      const int k = q >> 3, nv = q & 7;
      *(uint4*)&Bs[cur][k * 64 + nv * 8] =
          *(const uint4*)(Bm + (size_t)(k0 + k) * N + n0 + nv * 8);
    }
#endif
    __syncthreads();

    Frag af[2];
#pragma unroll
    for (int mi = 0; mi < 2; ++mi) {
      const unsigned short* ar = &As[cur][(w * 32 + mi * 16 + ln) * 40];
      af[mi].u4[0] = *(const uint4*)(ar + 16 * g);
      af[mi].u4[1] = *(const uint4*)(ar + 16 * g + 8);
    }
#pragma unroll
    for (int nt = 0; nt < 4; ++nt) {
      Frag bf;
#pragma unroll
      for (int e = 0; e < 16; ++e)              // B-frag: k = e + 16g, n = lane
        bf.s[e] = Bs[cur][(e + 16 * g) * 64 + nt * 16 + ln];
#pragma unroll
      for (int mi = 0; mi < 2; ++mi)
        acc[mi][nt] = wmma_bf16(af[mi].v, bf.v, acc[mi][nt]);
    }
    __syncthreads();
  }

#pragma unroll
  for (int mi = 0; mi < 2; ++mi)
#pragma unroll
    for (int nt = 0; nt < 4; ++nt) {
      const int col = n0 + nt * 16 + ln;
      const float bv = bias[col];
#pragma unroll
      for (int r = 0; r < 8; ++r) {
        const int row = m0 + w * 32 + mi * 16 + r + 8 * g;
        float v = acc[mi][nt][r] + bv;
        if (RES)  v += residual[(size_t)row * N + col];
        if (RELU) v = fmaxf(v, 0.0f);
        if (OUTBF) ((unsigned short*)out)[(size_t)row * N + col] = f2bf(v);
        else       ((float*)out)[(size_t)row * N + col] = v;
      }
    }
}

// ---------------------------------------------------------------------------
// Flash attention: one wave per (b, h, 16-query tile).
// Scores computed TRANSPOSED (S^T = K_tile @ Q^T) so each lane owns one query
// column: online-softmax stats are in-lane reductions + ONE shfl_xor(16) each,
// instead of a 4-step bpermute tree per row. P packed to LDS as b128 stores;
// V staged with async global->LDS (ASYNCcnt) and consumed as B-fragments.
// ---------------------------------------------------------------------------
__global__ __launch_bounds__(256) void flash_attn(const unsigned short* __restrict__ Qb,
                                                  const unsigned short* __restrict__ Kb,
                                                  const unsigned short* __restrict__ Vb,
                                                  const int* __restrict__ mask,
                                                  unsigned short* __restrict__ ctx) {
  __shared__ unsigned short Pls[8][16][32];   // per-wave P (16 q x 32 k)
  __shared__ unsigned short Vls[8][32][72];   // per-wave V (32 k x 64 d, padded)
  __shared__ float          Stat[8][16];      // per-wave per-query broadcast

  const int w    = threadIdx.x >> 5;
  const int lane = threadIdx.x & 31;
  const int g    = lane >> 4;
  const int ln   = lane & 15;

  const int wg = blockIdx.x * 8 + w;
  const int qt = wg & 127;
  const int h  = (wg >> 7) & 15;
  const int b  = wg >> 11;
  const int q0 = qt * 16;

  // Q as B-matrix fragments (lane = query column; contraction over d).
  Frag qb[2];
  {
    const unsigned short* qp = Qb + ((size_t)b * SEQ + q0 + ln) * D_MODEL + h * DK;
#pragma unroll
    for (int dc = 0; dc < 2; ++dc) {
      qb[dc].u4[0] = *(const uint4*)(qp + dc * 32 + 16 * g);
      qb[dc].u4[1] = *(const uint4*)(qp + dc * 32 + 16 * g + 8);
    }
  }

  float mq = -3.0e38f, lq = 0.0f;   // running stats for this lane's query
  v8f o[4];
#pragma unroll
  for (int t = 0; t < 4; ++t) o[t] = {};

  for (int kc = 0; kc < SEQ / 32; ++kc) {
    const int s0k = kc * 32;

    // ---- stage V chunk (lane <-> key row, 128B contiguous) ----
    {
      const unsigned short* gv =
          Vb + ((size_t)b * SEQ + s0k + lane) * D_MODEL + h * DK;
#if HAS_ASYNC
#pragma unroll
      for (int j = 0; j < 8; ++j)
        __builtin_amdgcn_global_load_async_to_lds_b128(
            to_glb(gv + j * 8), to_lds(&Vls[w][lane][j * 8]), 0, 0);
#else
#pragma unroll
      for (int j = 0; j < 8; ++j)
        *(uint4*)&Vls[w][lane][j * 8] = *(const uint4*)(gv + j * 8);
#endif
    }

    // ---- S^T[key][query]: A = K tile rows, B = Q^T; scale + mask ----
    float sc[2][8];
#pragma unroll
    for (int nt = 0; nt < 2; ++nt) {
      v8f s = {};
      const unsigned short* kp =
          Kb + ((size_t)b * SEQ + s0k + nt * 16 + ln) * D_MODEL + h * DK;
#pragma unroll
      for (int dc = 0; dc < 2; ++dc) {
        Frag kf;                               // A-frag: row = key, k = d
        kf.u4[0] = *(const uint4*)(kp + dc * 32 + 16 * g);
        kf.u4[1] = *(const uint4*)(kp + dc * 32 + 16 * g + 8);
        s = wmma_bf16(kf.v, qb[dc].v, s);
      }
      union { int4 v; int i[4]; } mk0, mk1;    // mask per element-key r+8g
      mk0.v = *(const int4*)(mask + b * SEQ + s0k + nt * 16 + 8 * g);
      mk1.v = *(const int4*)(mask + b * SEQ + s0k + nt * 16 + 8 * g + 4);
#pragma unroll
      for (int r = 0; r < 8; ++r) {
        const int mv = (r < 4) ? mk0.i[r] : mk1.i[r - 4];
        sc[nt][r] = (mv != 0) ? s[r] * 0.125f : -1e9f;   // 1/sqrt(64)
      }
    }

    // ---- per-query online softmax: in-lane reduce + one shfl_xor(16) ----
    float pm = -3.0e38f;
#pragma unroll
    for (int nt = 0; nt < 2; ++nt)
#pragma unroll
      for (int r = 0; r < 8; ++r) pm = fmaxf(pm, sc[nt][r]);
    pm = fmaxf(pm, __shfl_xor(pm, 16));
    const float mn = fmaxf(mq, pm);
    const float aq = __expf(mq - mn);
    mq = mn;
    float ps = 0.0f;
#pragma unroll
    for (int nt = 0; nt < 2; ++nt)
#pragma unroll
      for (int r = 0; r < 8; ++r) {
        const float p = __expf(sc[nt][r] - mn);
        sc[nt][r] = p;
        ps += p;
      }
    ps += __shfl_xor(ps, 16);
    lq = lq * aq + ps;

    if (g == 0) Stat[w][ln] = aq;              // broadcast alpha to row layout

    // ---- pack P rows (query=ln) into LDS as two b128 stores ----
#pragma unroll
    for (int nt = 0; nt < 2; ++nt) {
      uint4 pk;
      pk.x = pack2(sc[nt][0], sc[nt][1]);
      pk.y = pack2(sc[nt][2], sc[nt][3]);
      pk.z = pack2(sc[nt][4], sc[nt][5]);
      pk.w = pack2(sc[nt][6], sc[nt][7]);
      *(uint4*)&Pls[w][ln][nt * 16 + 8 * g] = pk;   // keys nt*16+8g..+7
    }
    WAIT_DS();   // per-wave LDS pipe is in-order; drain before re-reading

    {
      const float4 a0 = *(const float4*)&Stat[w][8 * g];
      const float4 a1 = *(const float4*)&Stat[w][8 * g + 4];
      const float al[8] = {a0.x, a0.y, a0.z, a0.w, a1.x, a1.y, a1.z, a1.w};
#pragma unroll
      for (int t = 0; t < 4; ++t)
#pragma unroll
        for (int r = 0; r < 8; ++r) o[t][r] *= al[r];
    }

    Frag pf;                                   // P as A-matrix (16 q x 32 k)
    pf.u4[0] = *(const uint4*)&Pls[w][ln][16 * g];
    pf.u4[1] = *(const uint4*)&Pls[w][ln][16 * g + 8];

#if HAS_ASYNC
    WAIT_ASYNC();                              // V tile resident in LDS
#endif
#pragma unroll
    for (int nt = 0; nt < 4; ++nt) {
      Frag vf;                                 // B-frag: k = e+16g, n = dim
#pragma unroll
      for (int e = 0; e < 16; ++e)
        vf.s[e] = Vls[w][e + 16 * g][nt * 16 + ln];
      o[nt] = wmma_bf16(pf.v, vf.v, o[nt]);
    }
  }

  if (g == 0) Stat[w][ln] = 1.0f / lq;         // broadcast 1/l to row layout
  WAIT_DS();
  {
    const float4 i0 = *(const float4*)&Stat[w][8 * g];
    const float4 i1 = *(const float4*)&Stat[w][8 * g + 4];
    const float inv[8] = {i0.x, i0.y, i0.z, i0.w, i1.x, i1.y, i1.z, i1.w};
#pragma unroll
    for (int nt = 0; nt < 4; ++nt)
#pragma unroll
      for (int r = 0; r < 8; ++r) {
        const size_t idx =
            ((size_t)b * SEQ + q0 + r + 8 * g) * D_MODEL + h * DK + nt * 16 + ln;
        ctx[idx] = f2bf(o[nt][r] * inv[r]);
      }
  }
}

// ---------------------------------------------------------------------------
// Host-side orchestration
// ---------------------------------------------------------------------------
extern "C" void kernel_launch(void* const* d_in, const int* in_sizes, int n_in,
                              void* d_out, int out_size, void* d_ws, size_t ws_size,
                              hipStream_t stream) {
  const float* x     = (const float*)d_in[0];
  const int*   mask  = (const int*)d_in[1];
  const float* ln1_g = (const float*)d_in[2];
  const float* ln1_b = (const float*)d_in[3];
  const float* Wq    = (const float*)d_in[4];
  const float* bq    = (const float*)d_in[5];
  const float* Wk    = (const float*)d_in[6];
  const float* bk    = (const float*)d_in[7];
  const float* Wv    = (const float*)d_in[8];
  const float* bv    = (const float*)d_in[9];
  const float* Wo    = (const float*)d_in[10];
  const float* bo    = (const float*)d_in[11];
  const float* ln2_g = (const float*)d_in[12];
  const float* ln2_b = (const float*)d_in[13];
  const float* W1    = (const float*)d_in[14];
  const float* b1    = (const float*)d_in[15];
  const float* W2    = (const float*)d_in[16];
  const float* b2    = (const float*)d_in[17];

  char* ws = (char*)d_ws;
  auto carve = [&](size_t bytes) {
    char* p = ws;
    ws += (bytes + 255) & ~(size_t)255;
    return p;
  };
  unsigned short* wq_bf = (unsigned short*)carve((size_t)D_MODEL * D_MODEL * 2);
  unsigned short* wk_bf = (unsigned short*)carve((size_t)D_MODEL * D_MODEL * 2);
  unsigned short* wv_bf = (unsigned short*)carve((size_t)D_MODEL * D_MODEL * 2);
  unsigned short* wo_bf = (unsigned short*)carve((size_t)D_MODEL * D_MODEL * 2);
  unsigned short* w1_bf = (unsigned short*)carve((size_t)D_MODEL * FF_DIM * 2);
  unsigned short* w2_bf = (unsigned short*)carve((size_t)FF_DIM * D_MODEL * 2);
  unsigned short* xn_bf = (unsigned short*)carve((size_t)ROWS * D_MODEL * 2);  // ln1 then ln2
  unsigned short* q_bf  = (unsigned short*)carve((size_t)ROWS * D_MODEL * 2);
  unsigned short* k_bf  = (unsigned short*)carve((size_t)ROWS * D_MODEL * 2);
  unsigned short* v_bf  = (unsigned short*)carve((size_t)ROWS * D_MODEL * 2);
  unsigned short* c_bf  = (unsigned short*)carve((size_t)ROWS * D_MODEL * 2);
  float*          x_mid = (float*)carve((size_t)ROWS * D_MODEL * 4);
  unsigned short* ff_bf = (unsigned short*)carve((size_t)ROWS * FF_DIM * 2);

  // 1. weights -> bf16
  {
    const int nD = D_MODEL * D_MODEL, nF = D_MODEL * FF_DIM;
    cvt_f32_bf16<<<(nD + 255) / 256, 256, 0, stream>>>(Wq, wq_bf, nD);
    cvt_f32_bf16<<<(nD + 255) / 256, 256, 0, stream>>>(Wk, wk_bf, nD);
    cvt_f32_bf16<<<(nD + 255) / 256, 256, 0, stream>>>(Wv, wv_bf, nD);
    cvt_f32_bf16<<<(nD + 255) / 256, 256, 0, stream>>>(Wo, wo_bf, nD);
    cvt_f32_bf16<<<(nF + 255) / 256, 256, 0, stream>>>(W1, w1_bf, nF);
    cvt_f32_bf16<<<(nF + 255) / 256, 256, 0, stream>>>(W2, w2_bf, nF);
  }

  // 2. xn = LN1(x)
  layernorm_bf16<<<ROWS, 256, 0, stream>>>(x, ln1_g, ln1_b, xn_bf);

  // 3. Q/K/V projections
  {
    dim3 grid(D_MODEL / 64, ROWS / 128);
    gemm_bias<false, false, true><<<grid, 128, 0, stream>>>(
        xn_bf, wq_bf, bq, nullptr, q_bf, ROWS, D_MODEL, D_MODEL);
    gemm_bias<false, false, true><<<grid, 128, 0, stream>>>(
        xn_bf, wk_bf, bk, nullptr, k_bf, ROWS, D_MODEL, D_MODEL);
    gemm_bias<false, false, true><<<grid, 128, 0, stream>>>(
        xn_bf, wv_bf, bv, nullptr, v_bf, ROWS, D_MODEL, D_MODEL);
  }

  // 4. flash attention -> ctx
  flash_attn<<<(BATCH * HEADS * (SEQ / 16)) / 8, 256, 0, stream>>>(
      q_bf, k_bf, v_bf, mask, c_bf);

  // 5. x_mid = x + ctx @ Wo + bo
  {
    dim3 grid(D_MODEL / 64, ROWS / 128);
    gemm_bias<false, true, false><<<grid, 128, 0, stream>>>(
        c_bf, wo_bf, bo, x, x_mid, ROWS, D_MODEL, D_MODEL);
  }

  // 6. xn2 = LN2(x_mid)
  layernorm_bf16<<<ROWS, 256, 0, stream>>>(x_mid, ln2_g, ln2_b, xn_bf);

  // 7. ff = relu(xn2 @ W1 + b1)
  {
    dim3 grid(FF_DIM / 64, ROWS / 128);
    gemm_bias<true, false, true><<<grid, 128, 0, stream>>>(
        xn_bf, w1_bf, b1, nullptr, ff_bf, ROWS, FF_DIM, D_MODEL);
  }

  // 8. out = x_mid + ff @ W2 + b2
  {
    dim3 grid(D_MODEL / 64, ROWS / 128);
    gemm_bias<false, true, false><<<grid, 128, 0, stream>>>(
        ff_bf, w2_bf, b2, x_mid, (float*)d_out, ROWS, D_MODEL, FF_DIM);
  }

  (void)in_sizes; (void)n_in; (void)out_size; (void)ws_size;
}
